// EMAttention2d_16243566313761
// MI455X (gfx1250) — matrix operand order repaired and checked by reading, not compiled, not run on hardware
//
#include <hip/hip_runtime.h>

// ---------------------------------------------------------------------------
// EMAttention2d for MI455X (gfx1250): bf16 WMMA GEMM pipeline, f32 accumulate.
// B=16, C=512, H=W=64 (N=4096), K=64, NUM_ITER=3.
// Round 3: async global->LDS staging via GLOBAL_LOAD_ASYNC_TO_LDS_B128
// (v4i addrspace-qualified pointers per clang's builtin signature) with a
// double-buffered LDS pipeline; register-prefetch fallback otherwise.
// ---------------------------------------------------------------------------

typedef __attribute__((ext_vector_type(16))) __bf16 v16bf;
typedef __attribute__((ext_vector_type(8)))  float  v8f;
typedef __attribute__((ext_vector_type(4)))  int    v4i;

#define BDIM   16
#define CDIM   512
#define NPIX   4096
#define KDIM   64
#define NITER  3

#if defined(__AMDGCN__) && __has_builtin(__builtin_amdgcn_global_load_async_to_lds_b128) && __has_builtin(__builtin_amdgcn_s_wait_asynccnt)
#define USE_ASYNC_LDS 1
#else
#define USE_ASYNC_LDS 0
#endif

__device__ __forceinline__ unsigned short f2bf(float f) {
    unsigned u = __float_as_uint(f);
    unsigned r = u + 0x7fffu + ((u >> 16) & 1u);   // round-to-nearest-even
    return (unsigned short)(r >> 16);
}
__device__ __forceinline__ float bf2f(unsigned short s) {
    return __uint_as_float(((unsigned)s) << 16);
}

// ---------------------------------------------------------------------------
// small utility kernels
// ---------------------------------------------------------------------------
__global__ void cvt_f32_bf16_kernel(const float* __restrict__ in,
                                    unsigned short* __restrict__ out, long n) {
    long i = (long)blockIdx.x * blockDim.x + threadIdx.x;
    long stride = (long)gridDim.x * blockDim.x;
    for (; i < n; i += stride) out[i] = f2bf(in[i]);
}

__global__ void zero_f32_kernel(float* __restrict__ p, long n) {
    long i = (long)blockIdx.x * blockDim.x + threadIdx.x;
    long stride = (long)gridDim.x * blockDim.x;
    for (; i < n; i += stride) p[i] = 0.0f;
}

// broadcast initial codebook mu (C,K) f32 into per-batch mu_t (B,K,C) and
// mu_ck (B,C,K) bf16
__global__ void mu_init_kernel(const float* __restrict__ mu,
                               unsigned short* __restrict__ mu_t,
                               unsigned short* __restrict__ mu_ck) {
    int i = blockIdx.x * blockDim.x + threadIdx.x;       // over C*K
    if (i >= CDIM * KDIM) return;
    int c = i / KDIM, k = i % KDIM;
    unsigned short v = f2bf(mu[i]);
    for (int b = 0; b < BDIM; ++b) {
        mu_t[((size_t)b * KDIM + k) * CDIM + c] = v;
        mu_ck[((size_t)b * CDIM + c) * KDIM + k] = v;
    }
}

// ---------------------------------------------------------------------------
// Generic tiled WMMA GEMM: D(M,N) = A(M,Kc) * B(Kc,N)  (row-major, bf16 in,
// f32 accumulate).  Block tile 128x128, K-step 32, 8 waves of 64x32.
// Double-buffered LDS; async global->LDS staging when available.
// A-rows are clamped (not masked) so every wave issues a fixed number of
// async ops per tile; garbage rows only feed output rows >= M, never stored.
// ---------------------------------------------------------------------------
template <bool HAS_BIAS, bool OUT_BF16, bool WRITE_T, bool RELU, bool BN>
__global__ __launch_bounds__(256)
void gemm_wmma_kernel(const unsigned short* __restrict__ A, long aBatch,
                      const unsigned short* __restrict__ B, long bBatch,
                      float* __restrict__ Cf,
                      unsigned short* __restrict__ Cb, long cBatch,
                      unsigned short* __restrict__ Ct, long ctBatch,
                      const float* __restrict__ bias,
                      float* __restrict__ bnSum, float* __restrict__ bnSq,
                      int M, int N, int Kc) {
    const int bm = blockIdx.x * 128;
    const int bn = blockIdx.y * 128;
    const int bz = blockIdx.z;
    A += (size_t)bz * aBatch;
    B += (size_t)bz * bBatch;
    if (OUT_BF16) Cb += (size_t)bz * cBatch; else Cf += (size_t)bz * cBatch;
    if (WRITE_T)  Ct += (size_t)bz * ctBatch;

    __shared__ unsigned short As[2][128 * 40];  // 128 rows x 32 halfs (+pad)
    __shared__ unsigned short Bs[2][32 * 136];  // 32 rows x 128 halfs (+pad)
    __shared__ float rs[128];
    __shared__ float rq[128];

    const int tid  = threadIdx.x;
    const int lane = tid & 31;
    const int w    = tid >> 5;
    const int wr   = w >> 2;          // 0..1
    const int wc   = w & 3;           // 0..3
    const int wm0  = wr * 64;
    const int wn0  = wc * 32;
    const int lm   = lane & 15;
    const int lhi  = lane >> 4;

    if (BN && tid < 128) { rs[tid] = 0.0f; rq[tid] = 0.0f; }

    // ---- invariant per-thread staging coordinates -------------------------
    size_t aOff[2];  int aDst[2];
    size_t bOff[2];  int bDst[2];
#pragma unroll
    for (int i = 0; i < 2; ++i) {
        int vec  = tid * 2 + i;                 // 0..511
        int arow = vec >> 2, ak = (vec & 3) * 8;
        int growc = bm + arow; if (growc > M - 1) growc = M - 1;  // clamp
        aOff[i] = (size_t)growc * Kc + ak;      // +k0 at use
        aDst[i] = arow * 40 + ak;
        int brow = vec >> 4, bc = (vec & 15) * 8;
        bOff[i] = (size_t)brow * N + bn + bc;   // +k0*N at use
        bDst[i] = brow * 136 + bc;
    }

    union Frag { v16bf h; unsigned int u[8]; };
    v8f acc[4][2];
#pragma unroll
    for (int mi = 0; mi < 4; ++mi)
#pragma unroll
        for (int ni = 0; ni < 2; ++ni)
#pragma unroll
            for (int r = 0; r < 8; ++r) acc[mi][ni][r] = 0.0f;

    auto compute = [&](int buf) {
        Frag a[4], bb[2];
#pragma unroll
        for (int mi = 0; mi < 4; ++mi) {
            int row = wm0 + mi * 16 + lm;
#pragma unroll
            for (int v = 0; v < 8; ++v) {
                int kidx = (v & 3) * 2 + (v >> 2) * 16 + lhi * 8;
                a[mi].u[v] =
                    *(const unsigned int*)(&As[buf][row * 40 + kidx]);
            }
        }
#pragma unroll
        for (int ni = 0; ni < 2; ++ni) {
            int col = wn0 + ni * 16 + lm;
#pragma unroll
            for (int v = 0; v < 8; ++v) {
                int kk = (v & 3) * 2 + (v >> 2) * 16 + lhi * 8;
                unsigned lo = Bs[buf][kk * 136 + col];
                unsigned hi = Bs[buf][(kk + 1) * 136 + col];
                bb[ni].u[v] = lo | (hi << 16);
            }
        }
#pragma unroll
        for (int mi = 0; mi < 4; ++mi)
#pragma unroll
            for (int ni = 0; ni < 2; ++ni)
                acc[mi][ni] = __builtin_amdgcn_wmma_f32_16x16x32_bf16(
                    false, a[mi].h, false, bb[ni].h, (short)0, acc[mi][ni],
                    false, false);
    };

    const int nTiles = Kc >> 5;

#if USE_ASYNC_LDS
    auto stage = [&](int buf, int k0) {
#pragma unroll
        for (int i = 0; i < 2; ++i) {
            __builtin_amdgcn_global_load_async_to_lds_b128(
                (__attribute__((address_space(1))) v4i*)(A + aOff[i] + k0),
                (__attribute__((address_space(3))) v4i*)&As[buf][aDst[i]],
                0, 0);
            __builtin_amdgcn_global_load_async_to_lds_b128(
                (__attribute__((address_space(1))) v4i*)(B + bOff[i] +
                                                         (size_t)k0 * N),
                (__attribute__((address_space(3))) v4i*)&Bs[buf][bDst[i]],
                0, 0);
        }
    };

    stage(0, 0);
    for (int it = 0; it < nTiles; ++it) {
        const int cur = it & 1;
        if (it + 1 < nTiles) {
            stage(1 - cur, (it + 1) << 5);             // overlap DMA w/ math
            __builtin_amdgcn_s_wait_asynccnt(4);       // tile `cur` landed
        } else {
            __builtin_amdgcn_s_wait_asynccnt(0);
        }
        __syncthreads();
        compute(cur);
        __syncthreads();
    }
#else
    // register-prefetch double buffer fallback
    uint4 ra[2], rb[2];
    auto gload = [&](int k0) {
#pragma unroll
        for (int i = 0; i < 2; ++i) {
            ra[i] = *(const uint4*)(A + aOff[i] + k0);
            rb[i] = *(const uint4*)(B + bOff[i] + (size_t)k0 * N);
        }
    };
    gload(0);
    for (int it = 0; it < nTiles; ++it) {
        const int cur = it & 1;
#pragma unroll
        for (int i = 0; i < 2; ++i) {
            *(uint4*)(&As[cur][aDst[i]]) = ra[i];
            *(uint4*)(&Bs[cur][bDst[i]]) = rb[i];
        }
        if (it + 1 < nTiles) gload((it + 1) << 5);     // prefetch next tile
        __syncthreads();
        compute(cur);
        __syncthreads();
    }
#endif

    // -------- epilogue -----------------------------------------------------
#pragma unroll
    for (int mi = 0; mi < 4; ++mi) {
        int mbase = wm0 + mi * 16 + lhi * 8;   // local row base (8 rows/lane)
#pragma unroll
        for (int ni = 0; ni < 2; ++ni) {
            int gn = bn + wn0 + ni * 16 + lm;
            union { unsigned short s[8]; uint4 q; } tp;
#pragma unroll
            for (int r = 0; r < 8; ++r) {
                int lrow = mbase + r;
                int gm = bm + lrow;
                float v = acc[mi][ni][r];
                if (gm < M) {
                    if (HAS_BIAS) v += bias[gm];
                    if (RELU) v = fmaxf(v, 0.0f);
                    if (OUT_BF16) Cb[(size_t)gm * N + gn] = f2bf(v);
                    else          Cf[(size_t)gm * N + gn] = v;
                    if (BN) { atomicAdd(&rs[lrow], v); atomicAdd(&rq[lrow], v * v); }
                }
                tp.s[r] = f2bf(v);
            }
            if (WRITE_T && (bm + mbase) < M)
                *(uint4*)(Ct + (size_t)gn * M + bm + mbase) = tp.q;
        }
    }
    if (BN) {
        __syncthreads();
        if (tid < 128 && (bm + tid) < M) {
            atomicAdd(&bnSum[bm + tid], rs[tid]);
            atomicAdd(&bnSq[bm + tid], rq[tid]);
        }
    }
}

// ---------------------------------------------------------------------------
// Column softmax over k (64) for z_t (K,N) logits; writes z in both layouts
// and accumulates per-k sums over n (for the zn normalization).
// ---------------------------------------------------------------------------
__global__ __launch_bounds__(128)
void softmax_kernel(const float* __restrict__ zlog,
                    unsigned short* __restrict__ z_t,
                    unsigned short* __restrict__ z_nk,
                    float* __restrict__ zsum) {
    const int b = blockIdx.y;
    const int n = blockIdx.x * 128 + threadIdx.x;
    const int t = threadIdx.x;
    __shared__ float sm[64][129];

    const float* zl = zlog + (size_t)b * KDIM * NPIX;
    float mx = -1e30f;
#pragma unroll
    for (int k = 0; k < KDIM; ++k) {
        float v = zl[(size_t)k * NPIX + n];
        sm[k][t] = v;
        mx = fmaxf(mx, v);
    }
    float s = 0.0f;
#pragma unroll
    for (int k = 0; k < KDIM; ++k) {
        float e = __expf(sm[k][t] - mx);
        sm[k][t] = e;
        s += e;
    }
    float inv = 1.0f / s;
    union { unsigned short s16[64]; uint4 q[8]; } row;
#pragma unroll
    for (int k = 0; k < KDIM; ++k) {
        float z = sm[k][t] * inv;
        sm[k][t] = z;
        unsigned short bz = f2bf(z);
        z_t[((size_t)b * KDIM + k) * NPIX + n] = bz;
        row.s16[k] = bz;
    }
    uint4* dst = (uint4*)(z_nk + ((size_t)b * NPIX + n) * KDIM);
#pragma unroll
    for (int q = 0; q < 8; ++q) dst[q] = row.q[q];

    __syncthreads();
    if (t < KDIM) {
        float ss = 0.0f;
#pragma unroll 8
        for (int j = 0; j < 128; ++j) ss += sm[t][j];
        atomicAdd(&zsum[b * KDIM + t], ss);
    }
}

// ---------------------------------------------------------------------------
// mu update epilogue: zn rescale (1/(1e-6+colsum)) then L2-normalize along C;
// writes mu in both (K,C) and (C,K) bf16 layouts.
// ---------------------------------------------------------------------------
__global__ __launch_bounds__(128)
void mu_norm_kernel(const float* __restrict__ mu_raw,
                    const float* __restrict__ zsum,
                    unsigned short* __restrict__ mu_t,
                    unsigned short* __restrict__ mu_ck) {
    const int b = blockIdx.y, k = blockIdx.x, t = threadIdx.x;
    const float* row = mu_raw + ((size_t)b * KDIM + k) * CDIM;
    float inv = 1.0f / (1e-6f + zsum[b * KDIM + k]);
    __shared__ float red[128];
    float ss = 0.0f;
    for (int c = t; c < CDIM; c += 128) { float v = row[c] * inv; ss += v * v; }
    red[t] = ss;
    __syncthreads();
    for (int o = 64; o > 0; o >>= 1) {
        if (t < o) red[t] += red[t + o];
        __syncthreads();
    }
    float norm = sqrtf(red[0]);
    float r = inv / fmaxf(norm, 1e-12f);
    for (int c = t; c < CDIM; c += 128) {
        unsigned short v = f2bf(row[c] * r);
        mu_t[((size_t)b * KDIM + k) * CDIM + c] = v;
        mu_ck[((size_t)b * CDIM + c) * KDIM + k] = v;
    }
}

// ---------------------------------------------------------------------------
// BatchNorm finalize: per-channel scale/shift from accumulated sum/sumsq.
// ---------------------------------------------------------------------------
__global__ void bn_finalize_kernel(const float* __restrict__ bnSum,
                                   const float* __restrict__ bnSq,
                                   const float* __restrict__ gamma,
                                   const float* __restrict__ beta,
                                   float* __restrict__ scale,
                                   float* __restrict__ shift, float invCnt) {
    int c = blockIdx.x * blockDim.x + threadIdx.x;
    if (c >= CDIM) return;
    float m = bnSum[c] * invCnt;
    float var = bnSq[c] * invCnt - m * m;
    float sc = gamma[c] * rsqrtf(var + 1e-5f);
    scale[c] = sc;
    shift[c] = beta[c] - m * sc;
}

// ---------------------------------------------------------------------------
// out = relu(bn(h) + x)
// ---------------------------------------------------------------------------
__global__ void fuse_out_kernel(const unsigned short* __restrict__ h,
                                const float* __restrict__ x,
                                const float* __restrict__ scale,
                                const float* __restrict__ shift,
                                float* __restrict__ out, long total) {
    long i = (long)blockIdx.x * blockDim.x + threadIdx.x;
    long stride = (long)gridDim.x * blockDim.x;
    for (; i < total; i += stride) {
        int c = (int)((i / NPIX) % CDIM);
        float v = bf2f(h[i]) * scale[c] + shift[c] + x[i];
        out[i] = fmaxf(v, 0.0f);
    }
}

// ---------------------------------------------------------------------------
// host launcher
// ---------------------------------------------------------------------------
extern "C" void kernel_launch(void* const* d_in, const int* in_sizes, int n_in,
                              void* d_out, int out_size, void* d_ws,
                              size_t ws_size, hipStream_t stream) {
    const float* x      = (const float*)d_in[0];
    const float* mu     = (const float*)d_in[1];
    const float* stem_w = (const float*)d_in[2];
    const float* stem_b = (const float*)d_in[3];
    const float* head_w = (const float*)d_in[4];
    const float* head_b = (const float*)d_in[5];
    const float* gamma  = (const float*)d_in[6];
    const float* beta   = (const float*)d_in[7];
    float* out = (float*)d_out;

    const long BCN = (long)BDIM * CDIM * NPIX;      // 33.55M elements
    const long BKN = (long)BDIM * KDIM * NPIX;
    const long BKC = (long)BDIM * KDIM * CDIM;

    char* w = (char*)d_ws;
    size_t off = 0;
    auto take = [&](size_t bytes) -> char* {
        char* p = w + off;
        off = (off + bytes + 255) & ~(size_t)255;
        return p;
    };
    unsigned short* x_bf  = (unsigned short*)take(BCN * 2);
    unsigned short* y_cn  = (unsigned short*)take(BCN * 2);
    unsigned short* y_nc  = (unsigned short*)take(BCN * 2);
    unsigned short* y2r   = (unsigned short*)take(BCN * 2);
    unsigned short* h_bf  = (unsigned short*)take(BCN * 2);
    float*          zlog  = (float*)take(BKN * 4);
    unsigned short* z_t   = (unsigned short*)take(BKN * 2);
    unsigned short* z_nk  = (unsigned short*)take(BKN * 2);
    unsigned short* mu_t  = (unsigned short*)take(BKC * 2);
    unsigned short* mu_ck = (unsigned short*)take(BKC * 2);
    float*          mu_rw = (float*)take(BKC * 4);
    float*          zsum  = (float*)take((size_t)BDIM * KDIM * 4);
    unsigned short* sw_bf = (unsigned short*)take((size_t)CDIM * CDIM * 2);
    unsigned short* hw_bf = (unsigned short*)take((size_t)CDIM * CDIM * 2);
    float*          bnSum = (float*)take(CDIM * 4);
    float*          bnSq  = (float*)take(CDIM * 4);
    float*          bnSc  = (float*)take(CDIM * 4);
    float*          bnSh  = (float*)take(CDIM * 4);

    // ---- cast inputs to bf16 ----
    cvt_f32_bf16_kernel<<<4096, 256, 0, stream>>>(x, x_bf, BCN);
    cvt_f32_bf16_kernel<<<512, 256, 0, stream>>>(stem_w, sw_bf, (long)CDIM * CDIM);
    cvt_f32_bf16_kernel<<<512, 256, 0, stream>>>(head_w, hw_bf, (long)CDIM * CDIM);
    mu_init_kernel<<<(CDIM * KDIM + 255) / 256, 256, 0, stream>>>(mu, mu_t, mu_ck);

    // ---- stem: y = stem_w @ x + b  ->  y_cn (C,N) and y_nc (N,C) ----
    gemm_wmma_kernel<true, true, true, false, false>
        <<<dim3(4, 32, BDIM), 256, 0, stream>>>(
            sw_bf, 0, x_bf, (long)CDIM * NPIX, nullptr, y_cn, (long)CDIM * NPIX,
            y_nc, (long)NPIX * CDIM, stem_b, nullptr, nullptr,
            CDIM, NPIX, CDIM);

    // ---- EM iterations ----
    for (int it = 0; it < NITER; ++it) {
        zero_f32_kernel<<<4, 256, 0, stream>>>(zsum, (long)BDIM * KDIM);

        // logits: z_t(K,N) = mu_t(K,C) @ y_cn(C,N)
        gemm_wmma_kernel<false, false, false, false, false>
            <<<dim3(1, 32, BDIM), 256, 0, stream>>>(
                mu_t, (long)KDIM * CDIM, y_cn, (long)CDIM * NPIX, zlog, nullptr,
                (long)KDIM * NPIX, nullptr, 0, nullptr, nullptr, nullptr,
                KDIM, NPIX, CDIM);

        softmax_kernel<<<dim3(NPIX / 128, BDIM), 128, 0, stream>>>(
            zlog, z_t, z_nk, zsum);

        // mu_raw(K,C) = z_t(K,N) @ y_nc(N,C)
        gemm_wmma_kernel<false, false, false, false, false>
            <<<dim3(1, 4, BDIM), 256, 0, stream>>>(
                z_t, (long)KDIM * NPIX, y_nc, (long)NPIX * CDIM, mu_rw, nullptr,
                (long)KDIM * CDIM, nullptr, 0, nullptr, nullptr, nullptr,
                KDIM, CDIM, NPIX);

        mu_norm_kernel<<<dim3(KDIM, BDIM), 128, 0, stream>>>(mu_rw, zsum, mu_t,
                                                             mu_ck);
    }

    // ---- y2 = relu(mu_ck(C,K) @ z_t(K,N)) ----
    gemm_wmma_kernel<false, true, false, true, false>
        <<<dim3(4, 32, BDIM), 256, 0, stream>>>(
            mu_ck, (long)CDIM * KDIM, z_t, (long)KDIM * NPIX, nullptr, y2r,
            (long)CDIM * NPIX, nullptr, 0, nullptr, nullptr, nullptr,
            CDIM, NPIX, KDIM);

    // ---- head: h = head_w @ y2 + b, with fused BN partial sums ----
    zero_f32_kernel<<<1, 256, 0, stream>>>(bnSum, CDIM);
    zero_f32_kernel<<<1, 256, 0, stream>>>(bnSq, CDIM);
    gemm_wmma_kernel<true, true, false, false, true>
        <<<dim3(4, 32, BDIM), 256, 0, stream>>>(
            hw_bf, 0, y2r, (long)CDIM * NPIX, nullptr, h_bf, (long)CDIM * NPIX,
            nullptr, 0, head_b, bnSum, bnSq,
            CDIM, NPIX, CDIM);

    bn_finalize_kernel<<<2, 256, 0, stream>>>(bnSum, bnSq, gamma, beta, bnSc,
                                              bnSh, 1.0f / (float)(BDIM * NPIX));

    fuse_out_kernel<<<4096, 256, 0, stream>>>(h_bf, x, bnSc, bnSh, out, BCN);
}
